// Generator_82351702933707
// MI455X (gfx1250) — matrix operand order repaired and checked by reading, not compiled, not run on hardware
//
#include <hip/hip_runtime.h>
#include <math.h>
#include <stdint.h>

typedef __attribute__((ext_vector_type(16))) __bf16 v16bf;
typedef __attribute__((ext_vector_type(8)))  float  v8f;

// ---------------------------------------------------------------------------
// WMMA fragment index maps (CDNA5 ISA 7.12.2, wave32)
// A 16x32 bf16: lane L -> M = L%16; element j (VGPR pair p=j/2, lo=j&1):
//   p in [0,3]: K = 8*half + 2p + lo ; p in [4,7]: K = 16 + 8*half + 2(p-4) + lo
// B 32x16 bf16: lane L -> N = L%16; element j uses the same K map as A
// C/D f32 16x16: lane L -> N = L%16; VGPR r -> M = 8*half + r
// ---------------------------------------------------------------------------
__device__ __forceinline__ int a_k_of(int half, int j) {
    int p = j >> 1, lo = j & 1;
    return (p < 4) ? (half * 8 + 2 * p + lo) : (16 + half * 8 + 2 * (p - 4) + lo);
}
__device__ __forceinline__ int b_k_of(int half, int j) {
    int p = j >> 1, lo = j & 1;
    return (p < 4) ? (half * 8 + 2 * p + lo) : (16 + half * 8 + 2 * (p - 4) + lo);
}
__device__ __forceinline__ float leaky(float v) { return v >= 0.f ? v : 0.2f * v; }

// ---------------------------------------------------------------------------
// normalize: w0[b,:] = z[b,:] / max(||z[b,:]||, 1e-12)
// ---------------------------------------------------------------------------
__global__ void normalize_kernel(const float* __restrict__ z, float* __restrict__ out) {
    __shared__ float red[256];
    int b = blockIdx.x, t = threadIdx.x;
    float s = 0.f;
    for (int k = t; k < 512; k += 256) { float v = z[b * 512 + k]; s += v * v; }
    red[t] = s; __syncthreads();
    for (int off = 128; off > 0; off >>= 1) {
        if (t < off) red[t] += red[t + off];
        __syncthreads();
    }
    float inv = 1.f / fmaxf(sqrtf(red[0]), 1e-12f);
    for (int k = t; k < 512; k += 256) out[b * 512 + k] = z[b * 512 + k] * inv;
}

// ---------------------------------------------------------------------------
// Equalized linear via WMMA: y[16,N] = x[16,K] @ (w[N,K]*c)^T + bias, opt leaky
// One wave per 16-column tile. K multiple of 32.
// ---------------------------------------------------------------------------
__global__ __launch_bounds__(32)
void eq_gemm_wmma_kernel(const float* __restrict__ x, const float* __restrict__ w,
                         const float* __restrict__ bias, float* __restrict__ y,
                         int N, int K, float cscale, int act) {
    int lane = threadIdx.x & 31;
    int hf = lane >> 4, l = lane & 15;
    int n0 = blockIdx.x * 16;
    v8f acc = {};
    for (int kc = 0; kc < K; kc += 32) {
        v16bf A, B;
#pragma unroll
        for (int j = 0; j < 16; ++j) {
            A[j] = (__bf16)x[(size_t)l * K + kc + a_k_of(hf, j)];
            B[j] = (__bf16)w[(size_t)(n0 + l) * K + kc + b_k_of(hf, j)];
        }
        acc = __builtin_amdgcn_wmma_f32_16x16x32_bf16(false, A, false, B, (short)0, acc,
                                                      false, false);
    }
    int n = n0 + l;
#pragma unroll
    for (int r = 0; r < 8; ++r) {
        int m = hf * 8 + r;
        float v = acc[r] * cscale + bias[n];
        if (act) v = leaky(v);
        y[(size_t)m * N + n] = v;
    }
}

// ---------------------------------------------------------------------------
// demod: dd[b,o] = c * rsqrt(c^2 * sum_{i,k}(w[o,i,k]^2 * s[b,i]^2) + 1e-8),
// c = 1/sqrt(I*9). Folds the EqualizedWeight scale into the row factor.
// ---------------------------------------------------------------------------
__global__ void demod_kernel(const float* __restrict__ w, const float* __restrict__ s,
                             float* __restrict__ dd, int O, int I) {
    int idx = blockIdx.x * blockDim.x + threadIdx.x;
    if (idx >= 16 * O) return;
    int b = idx / O, o = idx % O;
    float c = rsqrtf((float)(I * 9));
    const float* wr = w + (size_t)o * I * 9;
    const float* sr = s + (size_t)b * I;
    float sum = 0.f;
    for (int i = 0; i < I; ++i) {
        float ws = 0.f;
#pragma unroll
        for (int t = 0; t < 9; ++t) { float ww = wr[i * 9 + t]; ws += ww * ww; }
        float si = sr[i];
        sum += ws * si * si;
    }
    dd[idx] = c * rsqrtf(c * c * sum + 1e-8f);
}

// ---------------------------------------------------------------------------
// modulate + cast to bf16: xm[b,i,p] = bf16(x[.,i,p] * s[b,i]); bcast=1 reads a
// batch-broadcast constant [I,HW].
// ---------------------------------------------------------------------------
__global__ void modcast_kernel(const float* __restrict__ x, const float* __restrict__ s,
                               __bf16* __restrict__ xm, int I, int HW, int bcast) {
    size_t idx = (size_t)blockIdx.x * blockDim.x + threadIdx.x;
    size_t total = (size_t)16 * I * HW;
    if (idx >= total) return;
    int p = (int)(idx % HW);
    size_t t = idx / HW;
    int i = (int)(t % I), b = (int)(t / I);
    float xv = bcast ? x[(size_t)i * HW + p] : x[idx];
    xm[idx] = (__bf16)(xv * s[(size_t)b * I + i]);
}

// ---------------------------------------------------------------------------
// bilinear (half-pixel, scale 2) sample with clamped borders
// ---------------------------------------------------------------------------
__device__ __forceinline__ float bilerp2x(const float* __restrict__ ch, int H, int W,
                                          int oy, int ox) {
    float fy = oy * 0.5f - 0.25f, fx = ox * 0.5f - 0.25f;
    int y0 = (int)floorf(fy), x0 = (int)floorf(fx);
    float wy = fy - y0, wx = fx - x0;
    int y0c = min(max(y0, 0), H - 1), y1c = min(max(y0 + 1, 0), H - 1);
    int x0c = min(max(x0, 0), W - 1), x1c = min(max(x0 + 1, 0), W - 1);
    float a = ch[y0c * W + x0c], b = ch[y0c * W + x1c];
    float c = ch[y1c * W + x0c], d = ch[y1c * W + x1c];
    return (a * (1.f - wx) + b * wx) * (1.f - wy) + (c * (1.f - wx) + d * wx) * wy;
}

__device__ __forceinline__ float up_smooth_at(const float* __restrict__ ch, int H, int W,
                                              int H2, int W2, int oy, int ox) {
    float acc = 0.f;
#pragma unroll
    for (int a = -1; a <= 1; ++a) {
        int yy = min(max(oy + a, 0), H2 - 1);
        float wy = (a == 0) ? 0.5f : 0.25f;
#pragma unroll
        for (int bb = -1; bb <= 1; ++bb) {
            int xx = min(max(ox + bb, 0), W2 - 1);
            acc += wy * ((bb == 0) ? 0.5f : 0.25f) * bilerp2x(ch, H, W, yy, xx);
        }
    }
    return acc;
}

// fused: upsample 2x (bilinear) -> 3x3 [1,2,1]x[1,2,1]/16 edge-padded blur ->
// per-channel style modulation -> bf16 cast. Never materializes fp32 x_up.
__global__ void up_smooth_modcast_kernel(const float* __restrict__ x,
                                         const float* __restrict__ s,
                                         __bf16* __restrict__ xm, int I, int H, int W) {
    int H2 = 2 * H, W2 = 2 * W, HW2 = H2 * W2;
    size_t idx = (size_t)blockIdx.x * blockDim.x + threadIdx.x;
    size_t total = (size_t)16 * I * HW2;
    if (idx >= total) return;
    int p = (int)(idx % HW2);
    size_t t = idx / HW2;
    int i = (int)(t % I), b = (int)(t / I);
    const float* ch = x + ((size_t)b * I + i) * H * W;
    float v = up_smooth_at(ch, H, W, H2, W2, p / W2, p % W2);
    xm[idx] = (__bf16)(v * s[(size_t)b * I + i]);
}

// same, fp32 out, no modulation (RGB path)
__global__ void up_smooth_f32_kernel(const float* __restrict__ x, float* __restrict__ y,
                                     int C, int H, int W) {
    int H2 = 2 * H, W2 = 2 * W, HW2 = H2 * W2;
    size_t idx = (size_t)blockIdx.x * blockDim.x + threadIdx.x;
    size_t total = (size_t)16 * C * HW2;
    if (idx >= total) return;
    int p = (int)(idx % HW2);
    size_t t = idx / HW2;
    int c = (int)(t % C), b = (int)(t / C);
    const float* ch = x + ((size_t)b * C + c) * H * W;
    y[idx] = up_smooth_at(ch, H, W, H2, W2, p / W2, p % W2);
}

// ---------------------------------------------------------------------------
// Modulated 3x3 conv as implicit GEMM on WMMA bf16 pipes with async LDS staging.
// Block = 4 waves = one 16-pixel tile x 64 outputs (4 o-tiles) for one sample.
// Per 32-channel K-chunk the halo window [p0-W-2, p0+W+17] is DMA'd
// global->LDS with GLOBAL_LOAD_ASYNC_TO_LDS_B32 (ASYNCcnt), then all 4 waves
// consume it for all 9 taps via DS reads.
// The VDST operand is the low 32 bits of the generic pointer into sbuf (the
// flat->LDS address mapping), which also makes the LDS object
// escape into the asm so the DS loads cannot be folded away. sbuf is also
// visibly zero-initialized once per block.
//   acc[o,p] = sum_{tap,i} W[o,i,tap] * xm[b,i,p+tap]
//   y = leaky(acc * dd[b,o] + nscale*noise[b,p] + bias[o])
// ---------------------------------------------------------------------------
#define NPMAX 276  // 2*128 + 20 (largest halo window, even)

__global__ __launch_bounds__(128)
void conv3x3_mod_wmma_kernel(const __bf16* __restrict__ xm, const float* __restrict__ w,
                             const float* __restrict__ dd, const float* __restrict__ noise,
                             const float* __restrict__ nscale, const float* __restrict__ bias,
                             float* __restrict__ y, int O, int I, int H, int W) {
    __shared__ __bf16 sbuf[NPMAX * 32];
    int tid = threadIdx.x;
    int lane = tid & 31, wave = tid >> 5;
    int hf = lane >> 4, l = lane & 15;
    int HW = H * W;
    int p0 = blockIdx.x * 16;
    int o0 = (blockIdx.y * 4 + wave) * 16;
    int b = blockIdx.z;
    const __bf16* xb = xm + (size_t)b * I * HW;
    int base = p0 - W - 2;   // even (p0 mult of 16, W even)
    int NP = 2 * W + 20;     // even window length
    bool active = (o0 < O);

    // visible one-time zero init (32-bit DS stores) -> loads can't fold to undef
    unsigned* sb32 = (unsigned*)sbuf;
    for (int e = tid; e < NPMAX * 16; e += 128) sb32[e] = 0u;

    int pix = p0 + l;
    int py = pix / W, px = pix - py * W;
    int qn[9];
    bool inb[9];
#pragma unroll
    for (int tap = 0; tap < 9; ++tap) {
        int dy = tap / 3 - 1, dx = tap % 3 - 1;
        int yy = py + dy, xx = px + dx;
        inb[tap] = (pix < HW) && ((unsigned)yy < (unsigned)H) && ((unsigned)xx < (unsigned)W);
        qn[tap] = yy * W + xx - base;
    }

    v8f acc = {};
    int nhalf = NP >> 1;  // bf16 pairs per channel row
    for (int kc = 0; kc < I; kc += 32) {
        __syncthreads();  // prior chunk fully consumed / init done
        // ---- async DMA stage: 32 channels x NP pixels (paired b32) ----
        for (int e = tid; e < nhalf * 32; e += 128) {
            int k = e / nhalf;
            int q = (e - k * nhalf) * 2;
            int srcp = base + q;
            srcp = min(max(srcp, 0), HW - 2);  // clamp: OOB slots never consumed
            // low 32 bits of generic pointer == wave-relative LDS byte offset;
            // deriving it from &sbuf[...] also escapes sbuf into the asm.
            unsigned laddr = (unsigned)(uintptr_t)&sbuf[k * NPMAX + q];
            unsigned long long ga =
                (unsigned long long)(uintptr_t)(xb + (size_t)(kc + k) * HW + srcp);
            asm volatile("global_load_async_to_lds_b32 %0, %1, off"
                         :: "v"(laddr), "v"(ga)
                         : "memory");
        }
        asm volatile("s_wait_asynccnt 0" ::: "memory");
        __syncthreads();  // staged data visible to all waves
        // ---- consume: 9 taps, one WMMA each ----
        if (active) {
#pragma unroll
            for (int tap = 0; tap < 9; ++tap) {
                v16bf A, B;
#pragma unroll
                for (int j = 0; j < 16; ++j) {
                    int ka = kc + a_k_of(hf, j);
                    A[j] = (__bf16)w[((size_t)(o0 + l) * I + ka) * 9 + tap];
                    int kb = b_k_of(hf, j);
                    B[j] = inb[tap] ? sbuf[kb * NPMAX + qn[tap]] : (__bf16)0.0f;
                }
                acc = __builtin_amdgcn_wmma_f32_16x16x32_bf16(false, A, false, B, (short)0,
                                                              acc, false, false);
            }
        }
    }

    if (active && pix < HW) {
        float ns = nscale[0];
        float nz = noise[(size_t)b * HW + pix];
#pragma unroll
        for (int r = 0; r < 8; ++r) {
            int o = o0 + hf * 8 + r;
            float v = acc[r] * dd[(size_t)b * O + o] + ns * nz + bias[o];
            y[((size_t)b * O + o) * HW + pix] = leaky(v);
        }
    }
}

// ---------------------------------------------------------------------------
// to_rgb: 1x1 modulated conv (no demod), O=3: out = leaky(c*sum_i w*s*x + bias)
// ---------------------------------------------------------------------------
__global__ void torgb_kernel(const float* __restrict__ x, const float* __restrict__ s,
                             const float* __restrict__ cw, const float* __restrict__ bias,
                             float* __restrict__ out, int I, int HW, int add) {
    int idx = blockIdx.x * blockDim.x + threadIdx.x;
    int total = 16 * 3 * HW;
    if (idx >= total) return;
    int p = idx % HW;
    int o = (idx / HW) % 3;
    int b = idx / (3 * HW);
    float c = rsqrtf((float)I);
    const float* xb = x + (size_t)b * I * HW + p;
    const float* sr = s + (size_t)b * I;
    const float* wr = cw + (size_t)o * I;
    float acc = 0.f;
    for (int i = 0; i < I; ++i) acc += wr[i] * sr[i] * xb[(size_t)i * HW];
    float v = leaky(acc * c + bias[o]);
    out[idx] = add ? out[idx] + v : v;
}

__global__ void copy_kernel(const float* __restrict__ s, float* __restrict__ d, int n) {
    int i = blockIdx.x * blockDim.x + threadIdx.x;
    if (i < n) d[i] = s[i];
}
__global__ void tanh_kernel(const float* __restrict__ s, float* __restrict__ d, int n) {
    int i = blockIdx.x * blockDim.x + threadIdx.x;
    if (i < n) d[i] = tanhf(s[i]);
}

// ---------------------------------------------------------------------------
// Host orchestration
// ---------------------------------------------------------------------------
extern "C" void kernel_launch(void* const* d_in, const int* in_sizes, int n_in,
                              void* d_out, int out_size, void* d_ws, size_t ws_size,
                              hipStream_t stream) {
    (void)in_sizes; (void)n_in; (void)out_size; (void)ws_size;
    int cur = 0;
    auto nf = [&]() { return (const float*)d_in[cur++]; };

    const float* z = nf();
    const float* noise[11]; for (int i = 0; i < 11; ++i) noise[i] = nf();
    const float* map_w[4];  for (int i = 0; i < 4; ++i)  map_w[i] = nf();
    const float* map_b[4];  for (int i = 0; i < 4; ++i)  map_b[i] = nf();
    const float* cst = nf();
    struct SB   { const float *sw, *sb, *cw, *scale, *bias; };
    struct RGBP { const float *sw, *sb, *cw, *bias; };
    SB st0;   st0.sw = nf(); st0.sb = nf(); st0.cw = nf(); st0.scale = nf(); st0.bias = nf();
    RGBP rg0; rg0.sw = nf(); rg0.sb = nf(); rg0.cw = nf(); rg0.bias = nf();
    SB bs1[5], bs2[5]; RGBP brg[5];
    for (int i = 0; i < 5; ++i) {
        bs1[i].sw = nf(); bs1[i].sb = nf(); bs1[i].cw = nf(); bs1[i].scale = nf(); bs1[i].bias = nf();
        bs2[i].sw = nf(); bs2[i].sb = nf(); bs2[i].cw = nf(); bs2[i].scale = nf(); bs2[i].bias = nf();
        brg[i].sw = nf(); brg[i].sb = nf(); brg[i].cw = nf(); brg[i].bias = nf();
    }

    // workspace carve (~107 MB)
    char* wsb = (char*)d_ws;
    size_t off = 0;
    auto carve = [&](size_t bytes) {
        void* p = wsb + off;
        off += (bytes + 255) & ~(size_t)255;
        return p;
    };
    float*  WB0 = (float*)carve(16 * 512 * 4);
    float*  WB1 = (float*)carve(16 * 512 * 4);
    float*  SS1 = (float*)carve(16 * 512 * 4);
    float*  SS2 = (float*)carve(16 * 512 * 4);  // last write = final returned style
    float*  SR  = (float*)carve(16 * 512 * 4);
    float*  DD  = (float*)carve(16 * 512 * 4);
    float*  X0  = (float*)carve((size_t)8388608 * 4);    // 16x32x128x128 max fp32
    float*  X1  = (float*)carve((size_t)8388608 * 4);
    __bf16* XM  = (__bf16*)carve((size_t)16777216 * 2);  // 16x64x128x128 max bf16
    float*  R0  = (float*)carve((size_t)786432 * 4);     // 16x3x128x128
    float*  R1  = (float*)carve((size_t)786432 * 4);

    const float C512 = 0.044194173824159216f;  // 1/sqrt(512)

    // ---- mapping network ----
    normalize_kernel<<<16, 256, 0, stream>>>(z, WB0);
    for (int l = 0; l < 4; ++l) {
        const float* src = (l & 1) ? WB1 : WB0;
        float*       dst = (l & 1) ? WB0 : WB1;
        eq_gemm_wmma_kernel<<<32, 32, 0, stream>>>(src, map_w[l], map_b[l], dst,
                                                   512, 512, C512, 1);
    }
    const float* wlat = WB0;

    // ---- style block 0 @ 4x4 (I=O=512), const input broadcast over batch ----
    eq_gemm_wmma_kernel<<<32, 32, 0, stream>>>(wlat, st0.sw, st0.sb, SS1, 512, 512, C512, 0);
    demod_kernel<<<(16 * 512 + 255) / 256, 256, 0, stream>>>(st0.cw, SS1, DD, 512, 512);
    {
        size_t tot = (size_t)16 * 512 * 16;
        modcast_kernel<<<(unsigned)((tot + 255) / 256), 256, 0, stream>>>(cst, SS1, XM, 512, 16, 1);
    }
    conv3x3_mod_wmma_kernel<<<dim3(1, 8, 16), 128, 0, stream>>>(
        XM, st0.cw, DD, noise[0], st0.scale, st0.bias, X0, 512, 512, 4, 4);
    eq_gemm_wmma_kernel<<<32, 32, 0, stream>>>(wlat, rg0.sw, rg0.sb, SR, 512, 512, C512, 0);
    torgb_kernel<<<(16 * 3 * 16 + 255) / 256, 256, 0, stream>>>(X0, SR, rg0.cw, rg0.bias,
                                                                R0, 512, 16, 0);

    // ---- generator blocks ----
    const int FE[6] = {512, 512, 256, 128, 64, 32};
    int H = 4;
    float *xc = X0, *xo = X1, *rc = R0, *ro = R1;
    for (int i = 0; i < 5; ++i) {
        int ci = FE[i], co = FE[i + 1];
        int H2 = 2 * H, HW2 = H2 * H2;
        int gy = (co >= 64) ? co / 64 : 1;

        // s1: style, demod, fused up+blur+modulate+cast, WMMA conv -> xo
        eq_gemm_wmma_kernel<<<ci / 16, 32, 0, stream>>>(wlat, bs1[i].sw, bs1[i].sb, SS1,
                                                        ci, 512, C512, 0);
        demod_kernel<<<(16 * co + 255) / 256, 256, 0, stream>>>(bs1[i].cw, SS1, DD, co, ci);
        {
            size_t tot = (size_t)16 * ci * HW2;
            up_smooth_modcast_kernel<<<(unsigned)((tot + 255) / 256), 256, 0, stream>>>(
                xc, SS1, XM, ci, H, H);
        }
        conv3x3_mod_wmma_kernel<<<dim3((HW2 + 15) / 16, gy, 16), 128, 0, stream>>>(
            XM, bs1[i].cw, DD, noise[1 + 2 * i], bs1[i].scale, bs1[i].bias, xo, co, ci, H2, H2);

        // s2: style, demod, modulate+cast, WMMA conv -> xc
        eq_gemm_wmma_kernel<<<co / 16, 32, 0, stream>>>(wlat, bs2[i].sw, bs2[i].sb, SS2,
                                                        co, 512, C512, 0);
        demod_kernel<<<(16 * co + 255) / 256, 256, 0, stream>>>(bs2[i].cw, SS2, DD, co, co);
        {
            size_t tot = (size_t)16 * co * HW2;
            modcast_kernel<<<(unsigned)((tot + 255) / 256), 256, 0, stream>>>(
                xo, SS2, XM, co, HW2, 0);
        }
        conv3x3_mod_wmma_kernel<<<dim3((HW2 + 15) / 16, gy, 16), 128, 0, stream>>>(
            XM, bs2[i].cw, DD, noise[2 + 2 * i], bs2[i].scale, bs2[i].bias, xc, co, co, H2, H2);

        // rgb: upsample+blur old rgb, add to_rgb(xc)
        {
            size_t tot = (size_t)16 * 3 * HW2;
            up_smooth_f32_kernel<<<(unsigned)((tot + 255) / 256), 256, 0, stream>>>(rc, ro, 3, H, H);
            eq_gemm_wmma_kernel<<<co / 16, 32, 0, stream>>>(wlat, brg[i].sw, brg[i].sb, SR,
                                                            co, 512, C512, 0);
            torgb_kernel<<<(unsigned)((tot + 255) / 256), 256, 0, stream>>>(
                xc, SR, brg[i].cw, brg[i].bias, ro, co, HW2, 1);
        }
        { float* t = rc; rc = ro; ro = t; }
        H = H2;
    }

    // ---- outputs: concat(x[16,32,128,128], s[16,32], tanh(rgb[16,3,128,128])) ----
    float* out_x = (float*)d_out;
    float* out_s = out_x + 8388608;
    float* out_rgb = out_s + 512;
    copy_kernel<<<(8388608 + 255) / 256, 256, 0, stream>>>(xc, out_x, 8388608);
    copy_kernel<<<2, 256, 0, stream>>>(SS2, out_s, 512);
    tanh_kernel<<<(786432 + 255) / 256, 256, 0, stream>>>(rc, out_rgb, 786432);
}
